// Model_48129403519610
// MI455X (gfx1250) — matrix operand order repaired and checked by reading, not compiled, not run on hardware
//
#include <hip/hip_runtime.h>

typedef __attribute__((ext_vector_type(16))) _Float16 v16h;
typedef __attribute__((ext_vector_type(8)))  _Float16 v8h;
typedef __attribute__((ext_vector_type(8)))  float    v8f;

#define PADW 80   // 8 (left pad, keeps 16B-aligned LDS stores) + 64 + 3 halo + slack
#define PADH 73   // 3 + 64 + 3, extended to row 72 so k=49..63 reads stay in (zeroed) bounds
#define XOFS 5    // padded-coord column px stored at tile col px+5 (img col xx -> xx+8)

// Relative LDS element offset of patch element k (valid for k 0..63; k>=49 reads
// zeroed tile rows -- their B columns are zero so values are don't-care).
__device__ __forceinline__ int patch_off(int k) {
  int dy = k / 7;
  int dx = k - dy * 7;
  return dy * PADW + dx;
}

// Build transposed, zero-padded, mean-adjusted weights in LDS:
//   wlT[n*64 + k] = W[k,n] - colmean[n]  for n<8 && k<49, else 0   (n=0..15, k=0..63)
// W is staged through LDS once (cooperative load) to avoid unrolled global-load chains.
// Contains two __syncthreads(); caller must sync again before reading wlT.
__device__ __forceinline__ void build_wlT(_Float16* wlT, float* wraw, float* cmean,
                                          const float* __restrict__ W, int tid) {
  for (int i = tid; i < 49 * 8; i += 256) wraw[i] = W[i];
  __syncthreads();
  if (tid < 8) {
    float s = 0.f;
    #pragma clang loop unroll(disable)
    for (int k = 0; k < 49; ++k) s += wraw[k * 8 + tid];
    cmean[tid] = s * (1.0f / 49.0f);
  }
  __syncthreads();
  for (int i = tid; i < 16 * 64; i += 256) {
    int n = i >> 6, k = i & 63;
    float v = 0.f;
    if (n < 8 && k < 49) v = wraw[k * 8 + n] - cmean[n];
    wlT[i] = (_Float16)v;
  }
}

// ---------------- Stage 1: x[64,64,64] f32 -> maps1[512,64,64] f16 ----------------
__global__ __launch_bounds__(256) void pcanet_stage1(const float* __restrict__ x,
                                                     const float* __restrict__ W1,
                                                     _Float16* __restrict__ maps) {
  __shared__ __align__(16) _Float16 tile[PADH * PADW];
  __shared__ __align__(16) _Float16 wlT[16 * 64];
  __shared__ __align__(16) float    wraw[49 * 8];
  __shared__ float cmean[8];
  const int img = blockIdx.x;
  const int tid = threadIdx.x;

  for (int i = tid; i < PADH * PADW; i += 256) tile[i] = (_Float16)0.f;
  build_wlT(wlT, wraw, cmean, W1, tid);   // internal syncs also order tile zero vs. fill

  const float4* xi = (const float4*)(x + (size_t)img * 4096);
  for (int i = tid; i < 1024; i += 256) {
    float4 v = xi[i];
    int idx = i * 4;
    int yy = idx >> 6, xx = idx & 63;
    _Float16* dst = &tile[(yy + 3) * PADW + xx + 8];
    dst[0] = (_Float16)v.x; dst[1] = (_Float16)v.y;
    dst[2] = (_Float16)v.z; dst[3] = (_Float16)v.w;
  }
  __syncthreads();

  const int lane = tid & 31;
  const int wave = tid >> 5;
  const int h    = lane >> 4;
  const int nm   = lane & 15;

  // B fragments: element i of v16h = K = 16h+i (b1: +32) -> contiguous in wlT
  v16h b0 = *(const v16h*)&wlT[nm * 64 + 16 * h];
  v16h b1 = *(const v16h*)&wlT[nm * 64 + 16 * h + 32];

  // Loop-invariant per-lane A gather offsets (fully unrolled -> registers).
  // A layout: element i -> K = 16h+i (i=0..15); a1 adds 32.
  int offA0[16], offA1[16];
  #pragma unroll
  for (int i = 0; i < 8; ++i) {
    offA0[i]     = patch_off(16 * h + i);
    offA0[i + 8] = patch_off(16 * h + 8 + i);
    offA1[i]     = patch_off(32 + 16 * h + i);
    offA1[i + 8] = patch_off(40 + 16 * h + i);
  }

  for (int it = 0; it < 32; ++it) {
    int base = (it * 8 + wave) * 16;   // 16 consecutive pixels in one row
    int y = base >> 6, x0 = base & 63;
    const _Float16* p = tile + y * PADW + (x0 + nm) + XOFS;

    v16h a0, a1;
    #pragma unroll
    for (int i = 0; i < 16; ++i) { a0[i] = p[offA0[i]]; a1[i] = p[offA1[i]]; }

    v8f c = {};
    c = __builtin_amdgcn_wmma_f32_16x16x32_f16(false, a0, false, b0, (short)0, c, false, false);
    c = __builtin_amdgcn_wmma_f32_16x16x32_f16(false, a1, false, b1, (short)0, c, false, false);

    // D: lane holds channel nm, pixels base + 8h + r (r=0..7) -> 8 contiguous f16 = b128 store
    if (nm < 8) {
      v8h hc;
      #pragma unroll
      for (int r = 0; r < 8; ++r) hc[r] = (_Float16)c[r];
      _Float16* dst = maps + ((size_t)(nm * 64 + img)) * 4096 + base + 8 * h;
      *(v8h*)dst = hc;
    }
  }
}

// ---------------- Stage 2: maps1 f16 -> codes[512,64,64] u8 ----------------
__global__ __launch_bounds__(256) void pcanet_stage2(const _Float16* __restrict__ maps,
                                                     const float* __restrict__ W2,
                                                     unsigned char* __restrict__ codes) {
  __shared__ __align__(16) _Float16 tile[PADH * PADW];
  __shared__ __align__(16) _Float16 wlT[16 * 64];
  __shared__ __align__(16) float    wraw[49 * 8];
  __shared__ float cmean[8];
  const int img = blockIdx.x;
  const int tid = threadIdx.x;

  for (int i = tid; i < PADH * PADW; i += 256) tile[i] = (_Float16)0.f;
  build_wlT(wlT, wraw, cmean, W2, tid);

  const v8h* mi = (const v8h*)(maps + (size_t)img * 4096);
  for (int i = tid; i < 512; i += 256) {
    v8h v = mi[i];
    int idx = i * 8;
    int yy = idx >> 6, xx = idx & 63;
    *(v8h*)&tile[(yy + 3) * PADW + xx + 8] = v;
  }
  __syncthreads();

  const int lane = tid & 31;
  const int wave = tid >> 5;
  const int h    = lane >> 4;
  const int nm   = lane & 15;

  v16h b0 = *(const v16h*)&wlT[nm * 64 + 16 * h];
  v16h b1 = *(const v16h*)&wlT[nm * 64 + 16 * h + 32];

  int offA0[16], offA1[16];
  #pragma unroll
  for (int i = 0; i < 8; ++i) {
    offA0[i]     = patch_off(16 * h + i);
    offA0[i + 8] = patch_off(16 * h + 8 + i);
    offA1[i]     = patch_off(32 + 16 * h + i);
    offA1[i + 8] = patch_off(40 + 16 * h + i);
  }

  for (int it = 0; it < 32; ++it) {
    int base = (it * 8 + wave) * 16;
    int y = base >> 6, x0 = base & 63;
    const _Float16* p = tile + y * PADW + (x0 + nm) + XOFS;

    v16h a0, a1;
    #pragma unroll
    for (int i = 0; i < 16; ++i) { a0[i] = p[offA0[i]]; a1[i] = p[offA1[i]]; }

    v8f c = {};
    c = __builtin_amdgcn_wmma_f32_16x16x32_f16(false, a0, false, b0, (short)0, c, false, false);
    c = __builtin_amdgcn_wmma_f32_16x16x32_f16(false, a1, false, b1, (short)0, c, false, false);

    // sign-binarize: accumulator VGPR r holds pixel (base + 8h' + r), channel = lane&15.
    // ballot over lanes gives channel bits of pixels r (bits 0..7) and 8+r (bits 16..23).
    unsigned t0 = __builtin_amdgcn_ballot_w32(c[0] > 0.0f);
    unsigned t1 = __builtin_amdgcn_ballot_w32(c[1] > 0.0f);
    unsigned t2 = __builtin_amdgcn_ballot_w32(c[2] > 0.0f);
    unsigned t3 = __builtin_amdgcn_ballot_w32(c[3] > 0.0f);
    unsigned t4 = __builtin_amdgcn_ballot_w32(c[4] > 0.0f);
    unsigned t5 = __builtin_amdgcn_ballot_w32(c[5] > 0.0f);
    unsigned t6 = __builtin_amdgcn_ballot_w32(c[6] > 0.0f);
    unsigned t7 = __builtin_amdgcn_ballot_w32(c[7] > 0.0f);

    if (lane < 16) {                 // lane m writes code of pixel base+m
      int r  = lane & 7;
      int hh = lane >> 3;
      unsigned w = (r & 4) ? ((r & 2) ? ((r & 1) ? t7 : t6) : ((r & 1) ? t5 : t4))
                           : ((r & 2) ? ((r & 1) ? t3 : t2) : ((r & 1) ? t1 : t0));
      unsigned bytev = (w >> (hh * 16)) & 0xFFu;
      unsigned code  = __brev(bytev) >> 24;   // channel 0 -> bit 7 (weight 2^(7-l))
      codes[(size_t)img * 4096 + base + lane] = (unsigned char)code;
    }
  }
}

// ---------------- Stage 3: per-block histogram + entropy ----------------
__global__ __launch_bounds__(256) void pcanet_stage3(const unsigned char* __restrict__ codes,
                                                     float* __restrict__ out) {
  __shared__ unsigned bins[256];
  const int bid = blockIdx.x;      // bid = blk*512 + i2
  const int i2  = bid & 511;
  const int blk = bid >> 9;
  const int by  = blk / 7, bx = blk - by * 7;
  const int tid = threadIdx.x;

  bins[tid] = 0u;
  __syncthreads();

  const int ty = tid >> 4, tx = tid & 15;
  unsigned code = codes[(size_t)i2 * 4096 + (by * 8 + ty) * 64 + (bx * 8 + tx)];
  atomicAdd(&bins[code], 1u);
  __syncthreads();

  unsigned cnt = bins[tid];
  float p = (float)cnt * (1.0f / 256.0f);
  float e = (cnt > 0u) ? (-p * __log2f(p)) : 0.0f;
  out[((size_t)blk * 256 + tid) * 512 + i2] = e;
}

extern "C" void kernel_launch(void* const* d_in, const int* in_sizes, int n_in,
                              void* d_out, int out_size, void* d_ws, size_t ws_size,
                              hipStream_t stream) {
  (void)in_sizes; (void)n_in; (void)out_size; (void)ws_size;
  const float* x  = (const float*)d_in[0];   // [64,64,64]
  const float* W1 = (const float*)d_in[1];   // [49,8]
  const float* W2 = (const float*)d_in[2];   // [49,8]
  float* out = (float*)d_out;                // [49*256, 8, 64]

  _Float16*      maps  = (_Float16*)d_ws;                          // 4 MB
  unsigned char* codes = (unsigned char*)d_ws + (size_t)(4 << 20); // 2 MB

  pcanet_stage1<<<64, 256, 0, stream>>>(x, W1, maps);
  pcanet_stage2<<<512, 256, 0, stream>>>(maps, W2, codes);
  pcanet_stage3<<<49 * 512, 256, 0, stream>>>(codes, out);
}
